// GPT2FrozenStatePlugin_8246337208449
// MI455X (gfx1250) — compile-verified, matrix operand order checked by reading, not run on hardware
//
#include <hip/hip_runtime.h>

// ---------------------------------------------------------------------------
// Types for CDNA5 WMMA (wave32, 16x16x32 f16 -> f32)
// ---------------------------------------------------------------------------
typedef _Float16 f16;
typedef __attribute__((ext_vector_type(16))) _Float16     v16h;
typedef __attribute__((ext_vector_type(8)))  float        v8f;
typedef __attribute__((ext_vector_type(4)))  unsigned int u32x4;

union Frag16 { v16h h; u32x4 q[2]; };

__device__ __forceinline__ v8f wmma_f16(const Frag16& a, const Frag16& b, v8f c) {
  return __builtin_amdgcn_wmma_f32_16x16x32_f16(false, a.h, false, b.h, (short)0, c,
                                                false, false);
}

#define GB   8
#define GT   1024
#define GD   768
#define GH   12
#define GL   12
#define GNT  60
#define GM   (GB * GT)   // 8192 rows

// ---------------------------------------------------------------------------
// Sequential automaton scan: pre-states per (b, t)
// ---------------------------------------------------------------------------
__global__ void k_scan(const int* __restrict__ ids, const int* __restrict__ mul,
                       int* __restrict__ pre) {
  int b = threadIdx.x;
  if (b >= GB) return;
  int s = 0;  // ID_ID
  for (int t = 0; t < GT; ++t) {
    pre[b * GT + t] = s;
    s = mul[ids[b * GT + t] * GNT + s];
  }
}

// ---------------------------------------------------------------------------
// Gather state embeddings -> f16 A matrix [M, 128]
// ---------------------------------------------------------------------------
__global__ void k_gather_state(const int* __restrict__ pre, const float* __restrict__ se,
                               f16* __restrict__ Sg) {
  size_t i = (size_t)blockIdx.x * blockDim.x + threadIdx.x;
  if (i >= (size_t)GM * 128) return;
  int d = (int)(i & 127);
  size_t r = i >> 7;
  Sg[i] = (f16)se[(size_t)pre[r] * 128 + d];
}

// ---------------------------------------------------------------------------
// Weight f32 [K,N] -> f16 "Bt" layout [N,K]
// ---------------------------------------------------------------------------
__global__ void k_w_to_f16T(const float* __restrict__ W, f16* __restrict__ Bt,
                            int K, int N) {
  size_t i = (size_t)blockIdx.x * blockDim.x + threadIdx.x;
  if (i >= (size_t)K * N) return;
  int k = (int)(i % K);
  size_t n = i / K;
  Bt[i] = (f16)W[(size_t)k * N + n];
}

// ---------------------------------------------------------------------------
// Elementwise helpers
// ---------------------------------------------------------------------------
__global__ void k_f32_to_f16(const float* __restrict__ x, f16* __restrict__ y, size_t n) {
  size_t i = (size_t)blockIdx.x * blockDim.x + threadIdx.x;
  if (i < n) y[i] = (f16)x[i];
}
__global__ void k_resid_add(float* __restrict__ h, const float* __restrict__ t, size_t n) {
  size_t i = (size_t)blockIdx.x * blockDim.x + threadIdx.x;
  if (i < n) h[i] += t[i];
}
__global__ void k_gelu_f16(const float* __restrict__ x, f16* __restrict__ y, size_t n) {
  size_t i = (size_t)blockIdx.x * blockDim.x + threadIdx.x;
  if (i >= n) return;
  float v = x[i];
  float g = 0.5f * v * (1.0f + tanhf(0.7978845608028654f * (v + 0.044715f * v * v * v)));
  y[i] = (f16)g;
}
__global__ void k_gate_fuse(float* __restrict__ h, const float* __restrict__ t1,
                            const float* __restrict__ t2, const float* __restrict__ t3,
                            size_t n) {
  size_t i = (size_t)blockIdx.x * blockDim.x + threadIdx.x;
  if (i >= n) return;
  float g = 1.0f / (1.0f + __expf(-(t1[i] + t2[i])));
  h[i] += g * t3[i];
}
__global__ void k_embed(const int* __restrict__ ids, const float* __restrict__ tok,
                        const float* __restrict__ wpe, const f16* __restrict__ s16,
                        float* __restrict__ h) {
  size_t i = (size_t)blockIdx.x * blockDim.x + threadIdx.x;
  if (i >= (size_t)GM * GD) return;
  int d = (int)(i % GD);
  size_t bt = i / GD;
  int t = (int)(bt % GT);
  h[i] = tok[(size_t)ids[bt] * GD + d] + (float)s16[i] + wpe[(size_t)t * GD + d];
}

// ---------------------------------------------------------------------------
// LayerNorm (f32 in -> f16 out), one block per row
// ---------------------------------------------------------------------------
__global__ __launch_bounds__(256)
void k_ln_f16(const float* __restrict__ x, const float* __restrict__ g,
              const float* __restrict__ b, f16* __restrict__ out) {
  __shared__ float red[256];
  const float* xr = x + (size_t)blockIdx.x * GD;
  float s = 0.f;
  for (int i = threadIdx.x; i < GD; i += 256) s += xr[i];
  red[threadIdx.x] = s; __syncthreads();
  for (int st = 128; st > 0; st >>= 1) {
    if (threadIdx.x < st) red[threadIdx.x] += red[threadIdx.x + st];
    __syncthreads();
  }
  float mean = red[0] / GD; __syncthreads();
  float v = 0.f;
  for (int i = threadIdx.x; i < GD; i += 256) { float d = xr[i] - mean; v += d * d; }
  red[threadIdx.x] = v; __syncthreads();
  for (int st = 128; st > 0; st >>= 1) {
    if (threadIdx.x < st) red[threadIdx.x] += red[threadIdx.x + st];
    __syncthreads();
  }
  float rstd = rsqrtf(red[0] / GD + 1e-5f);
  for (int i = threadIdx.x; i < GD; i += 256)
    out[(size_t)blockIdx.x * GD + i] = (f16)((xr[i] - mean) * rstd * g[i] + b[i]);
}

// Final LN on last token of each batch row (f32 out)
__global__ __launch_bounds__(256)
void k_lnf_last(const float* __restrict__ h, const float* __restrict__ g,
                const float* __restrict__ b, float* __restrict__ out) {
  __shared__ float red[256];
  const float* xr = h + ((size_t)blockIdx.x * GT + (GT - 1)) * GD;
  float s = 0.f;
  for (int i = threadIdx.x; i < GD; i += 256) s += xr[i];
  red[threadIdx.x] = s; __syncthreads();
  for (int st = 128; st > 0; st >>= 1) {
    if (threadIdx.x < st) red[threadIdx.x] += red[threadIdx.x + st];
    __syncthreads();
  }
  float mean = red[0] / GD; __syncthreads();
  float v = 0.f;
  for (int i = threadIdx.x; i < GD; i += 256) { float d = xr[i] - mean; v += d * d; }
  red[threadIdx.x] = v; __syncthreads();
  for (int st = 128; st > 0; st >>= 1) {
    if (threadIdx.x < st) red[threadIdx.x] += red[threadIdx.x + st];
    __syncthreads();
  }
  float rstd = rsqrtf(red[0] / GD + 1e-5f);
  for (int i = threadIdx.x; i < GD; i += 256)
    out[(size_t)blockIdx.x * GD + i] = (xr[i] - mean) * rstd * g[i] + b[i];
}

// ---------------------------------------------------------------------------
// WMMA GEMM:  C[M,N] = A[M,K](f16) * W + bias,  weights prepacked as Bt[N,K] f16.
// Block = 4 waves; each wave computes a 32x64 strip (2 A-frags x 4 B-frags ->
// 8 WMMA per 32-wide K-step; every B fragment reused across two A fragments).
// Grid: (N/256, M/32).
// ---------------------------------------------------------------------------
__global__ __launch_bounds__(128)
void k_gemm_wmma(const f16* __restrict__ A, const f16* __restrict__ Bt,
                 const float* __restrict__ bias, float* __restrict__ C,
                 int N, int K) {
  const int lane  = threadIdx.x & 31;
  const int wave  = threadIdx.x >> 5;
  const int halfq = lane >> 4;      // 0 or 1
  const int nc    = lane & 15;
  const int m0    = blockIdx.y * 32;
  const int n0    = blockIdx.x * 256 + wave * 64;

  const f16* arow0 = A + (size_t)(m0 + nc) * K + halfq * 8;
  const f16* arow1 = arow0 + (size_t)16 * K;
  const f16* brow  = Bt + (size_t)(n0 + nc) * K + halfq * 8;

  v8f acc00 = {}, acc01 = {}, acc02 = {}, acc03 = {};
  v8f acc10 = {}, acc11 = {}, acc12 = {}, acc13 = {};
  for (int k0 = 0; k0 < K; k0 += 32) {
    Frag16 a0, a1, b0, b1, b2, b3;
    a0.q[0] = *(const u32x4*)(arow0 + k0);
    a0.q[1] = *(const u32x4*)(arow0 + k0 + 16);
    a1.q[0] = *(const u32x4*)(arow1 + k0);
    a1.q[1] = *(const u32x4*)(arow1 + k0 + 16);
    const f16* bp = brow + k0;
    b0.q[0] = *(const u32x4*)(bp);             b0.q[1] = *(const u32x4*)(bp + 16);
    b1.q[0] = *(const u32x4*)(bp + 16 * K);    b1.q[1] = *(const u32x4*)(bp + 16 * K + 16);
    b2.q[0] = *(const u32x4*)(bp + 32 * K);    b2.q[1] = *(const u32x4*)(bp + 32 * K + 16);
    b3.q[0] = *(const u32x4*)(bp + 48 * K);    b3.q[1] = *(const u32x4*)(bp + 48 * K + 16);
    acc00 = wmma_f16(a0, b0, acc00);
    acc10 = wmma_f16(a1, b0, acc10);
    acc01 = wmma_f16(a0, b1, acc01);
    acc11 = wmma_f16(a1, b1, acc11);
    acc02 = wmma_f16(a0, b2, acc02);
    acc12 = wmma_f16(a1, b2, acc12);
    acc03 = wmma_f16(a0, b3, acc03);
    acc13 = wmma_f16(a1, b3, acc13);
  }
  const float bv0 = bias[n0 + nc];
  const float bv1 = bias[n0 + 16 + nc];
  const float bv2 = bias[n0 + 32 + nc];
  const float bv3 = bias[n0 + 48 + nc];
#pragma unroll
  for (int r = 0; r < 8; ++r) {
    int m = m0 + r + 8 * halfq;
    float* crow = C + (size_t)m * N + n0 + nc;
    crow[0]  = acc00[r] + bv0;
    crow[16] = acc01[r] + bv1;
    crow[32] = acc02[r] + bv2;
    crow[48] = acc03[r] + bv3;
    float* crow2 = crow + (size_t)16 * N;
    crow2[0]  = acc10[r] + bv0;
    crow2[16] = acc11[r] + bv1;
    crow2[32] = acc12[r] + bv2;
    crow2[48] = acc13[r] + bv3;
  }
}

// ---------------------------------------------------------------------------
// QKV [M,2304] f32 -> Q,K f16 [B*H, T, 64], Vt f16 [B*H, 64, T]
// ---------------------------------------------------------------------------
__global__ void k_qkv_split(const float* __restrict__ qkv, f16* __restrict__ Q,
                            f16* __restrict__ Kk, f16* __restrict__ Vt) {
  size_t i = (size_t)blockIdx.x * blockDim.x + threadIdx.x;
  if (i >= (size_t)GM * 3 * GD) return;
  int c = (int)(i % (3 * GD));
  size_t bt = i / (3 * GD);
  int t = (int)(bt % GT);
  int b = (int)(bt / GT);
  int which = c / GD;
  int dc = c % GD;
  int h = dc >> 6;
  int d = dc & 63;
  f16 v = (f16)qkv[i];
  size_t bh = (size_t)(b * GH + h);
  if (which == 0)      Q [(bh * GT + t) * 64 + d] = v;
  else if (which == 1) Kk[(bh * GT + t) * 64 + d] = v;
  else                 Vt[(bh * 64 + d) * GT + t] = v;
}

// ---------------------------------------------------------------------------
// Flash attention: per wave one 16-row query tile of one (b,h).
// Key tiles of 32; S = Q K^T via 4 wmma; O += P V via 4 wmma; online softmax.
// Grid: (T/64, H, B), block = 128 (4 waves, each its own q-tile).
// ---------------------------------------------------------------------------
__global__ __launch_bounds__(128)
void k_flash_attn(const f16* __restrict__ Qm, const f16* __restrict__ Km,
                  const f16* __restrict__ Vt, f16* __restrict__ O) {
  __shared__ f16 pbuf[4][16][40];   // row stride 80B keeps 16B alignment
  const int lane  = threadIdx.x & 31;
  const int wave  = threadIdx.x >> 5;
  const int halfq = lane >> 4;
  const int nc    = lane & 15;
  const int b = blockIdx.z, h = blockIdx.y;
  const int q0 = (blockIdx.x * 4 + wave) * 16;
  const size_t bh = (size_t)(b * GH + h);
  const f16* Qb = Qm + bh * GT * 64;
  const f16* Kb = Km + bh * GT * 64;
  const f16* Vb = Vt + bh * 64 * GT;

  // Q a-fragments for d-steps 0..31 and 32..63
  const f16* qrow = Qb + (size_t)(q0 + nc) * 64 + halfq * 8;
  Frag16 qa0, qa1;
  qa0.q[0] = *(const u32x4*)(qrow);       qa0.q[1] = *(const u32x4*)(qrow + 16);
  qa1.q[0] = *(const u32x4*)(qrow + 32);  qa1.q[1] = *(const u32x4*)(qrow + 48);

  v8f o0 = {}, o1 = {}, o2 = {}, o3 = {};
  float mrun[8], lrun[8];
#pragma unroll
  for (int r = 0; r < 8; ++r) { mrun[r] = -3.0e38f; lrun[r] = 0.f; }

  for (int k0 = 0; k0 <= q0 + 15; k0 += 32) {
    v8f s0 = {}, s1 = {};
    {
      const f16* kr = Kb + (size_t)(k0 + nc) * 64 + halfq * 8;
      Frag16 kb;
      kb.q[0] = *(const u32x4*)(kr);       kb.q[1] = *(const u32x4*)(kr + 16);
      s0 = wmma_f16(qa0, kb, s0);
      kb.q[0] = *(const u32x4*)(kr + 32);  kb.q[1] = *(const u32x4*)(kr + 48);
      s0 = wmma_f16(qa1, kb, s0);
      const f16* kr2 = kr + 16 * 64;
      kb.q[0] = *(const u32x4*)(kr2);      kb.q[1] = *(const u32x4*)(kr2 + 16);
      s1 = wmma_f16(qa0, kb, s1);
      kb.q[0] = *(const u32x4*)(kr2 + 32); kb.q[1] = *(const u32x4*)(kr2 + 48);
      s1 = wmma_f16(qa1, kb, s1);
    }
    // scale + causal mask, tile row-max
    float sm0[8], sm1[8], rmax[8];
#pragma unroll
    for (int r = 0; r < 8; ++r) {
      int mg = q0 + r + 8 * halfq;
      float v0 = s0[r] * 0.125f; if (k0 + nc > mg)      v0 = -1e9f;
      float v1 = s1[r] * 0.125f; if (k0 + 16 + nc > mg) v1 = -1e9f;
      sm0[r] = v0; sm1[r] = v1;
      rmax[r] = fmaxf(v0, v1);
    }
#pragma unroll
    for (int off = 1; off < 16; off <<= 1)
#pragma unroll
      for (int r = 0; r < 8; ++r) rmax[r] = fmaxf(rmax[r], __shfl_xor(rmax[r], off, 32));

    float alpha[8], psum[8];
#pragma unroll
    for (int r = 0; r < 8; ++r) {
      float nm = fmaxf(mrun[r], rmax[r]);
      alpha[r] = __expf(mrun[r] - nm);
      mrun[r] = nm;
      float p0 = __expf(sm0[r] - nm);
      float p1 = __expf(sm1[r] - nm);
      sm0[r] = p0; sm1[r] = p1;
      psum[r] = p0 + p1;
    }
#pragma unroll
    for (int off = 1; off < 16; off <<= 1)
#pragma unroll
      for (int r = 0; r < 8; ++r) psum[r] += __shfl_xor(psum[r], off, 32);
#pragma unroll
    for (int r = 0; r < 8; ++r) {
      lrun[r] = lrun[r] * alpha[r] + psum[r];
      o0[r] *= alpha[r]; o1[r] *= alpha[r]; o2[r] *= alpha[r]; o3[r] *= alpha[r];
    }
    // transpose P (C layout -> A layout) via per-wave LDS tile
#pragma unroll
    for (int r = 0; r < 8; ++r) {
      pbuf[wave][r + 8 * halfq][nc]      = (f16)sm0[r];
      pbuf[wave][r + 8 * halfq][16 + nc] = (f16)sm1[r];
    }
    asm volatile("s_wait_dscnt 0" ::: "memory");  // same-wave DS RAW fence
    Frag16 pa;
    pa.q[0] = *(const u32x4*)&pbuf[wave][nc][halfq * 8];
    pa.q[1] = *(const u32x4*)&pbuf[wave][nc][16 + halfq * 8];

    const f16* vr = Vb + (size_t)nc * GT + k0 + halfq * 8;
    Frag16 vb;
    vb.q[0] = *(const u32x4*)(vr);                vb.q[1] = *(const u32x4*)(vr + 16);
    o0 = wmma_f16(pa, vb, o0);
    vb.q[0] = *(const u32x4*)(vr + 16 * GT);      vb.q[1] = *(const u32x4*)(vr + 16 * GT + 16);
    o1 = wmma_f16(pa, vb, o1);
    vb.q[0] = *(const u32x4*)(vr + 32 * GT);      vb.q[1] = *(const u32x4*)(vr + 32 * GT + 16);
    o2 = wmma_f16(pa, vb, o2);
    vb.q[0] = *(const u32x4*)(vr + 48 * GT);      vb.q[1] = *(const u32x4*)(vr + 48 * GT + 16);
    o3 = wmma_f16(pa, vb, o3);
  }
#pragma unroll
  for (int r = 0; r < 8; ++r) {
    float inv = 1.0f / lrun[r];
    int t = q0 + r + 8 * halfq;
    f16* orow = O + ((size_t)(b * GT + t) * GD) + h * 64 + nc;
    orow[0]  = (f16)(o0[r] * inv);
    orow[16] = (f16)(o1[r] * inv);
    orow[32] = (f16)(o2[r] * inv);
    orow[48] = (f16)(o3[r] * inv);
  }
}

// ---------------------------------------------------------------------------
// Head: logits[b,n] = lnf(h_last)[b,:] . head_w[:,n] + head_b[n]   (tiny, f32)
// ---------------------------------------------------------------------------
__global__ void k_head(const float* __restrict__ hf, const float* __restrict__ W,
                       const float* __restrict__ bb, float* __restrict__ out) {
  int i = (int)threadIdx.x;
  if (i >= GB * GNT) return;
  int b = i / GNT, n = i % GNT;
  const float* xr = hf + (size_t)b * GD;
  float acc = bb[n];
  for (int d = 0; d < GD; ++d) acc += xr[d] * W[(size_t)d * GNT + n];
  out[i] = acc;
}

// ---------------------------------------------------------------------------
// Host-side orchestration
// ---------------------------------------------------------------------------
static inline size_t alignup(size_t x) { return (x + 255) & ~(size_t)255; }

extern "C" void kernel_launch(void* const* d_in, const int* in_sizes, int n_in,
                              void* d_out, int out_size, void* d_ws, size_t ws_size,
                              hipStream_t stream) {
  const int*   input_ids = (const int*)d_in[0];
  const int*   mul       = (const int*)d_in[1];
  const float* tok_emb   = (const float*)d_in[2];
  const float* state_emb = (const float*)d_in[3];
  const float* sp_w  = (const float*)d_in[4];
  const float* sp_b  = (const float*)d_in[5];
  const float* wh_w  = (const float*)d_in[6];
  const float* wh_b  = (const float*)d_in[7];
  const float* ws_w  = (const float*)d_in[8];
  const float* ws_b  = (const float*)d_in[9];
  const float* wd_w  = (const float*)d_in[10];
  const float* wd_b  = (const float*)d_in[11];
  const float* head_w = (const float*)d_in[12];
  const float* head_b = (const float*)d_in[13];
  const float* wpe   = (const float*)d_in[14];
  const float* ln1_g = (const float*)d_in[15];
  const float* ln1_b = (const float*)d_in[16];
  const float* attn_w = (const float*)d_in[17];
  const float* attn_b = (const float*)d_in[18];
  const float* proj_w = (const float*)d_in[19];
  const float* proj_b = (const float*)d_in[20];
  const float* ln2_g = (const float*)d_in[21];
  const float* ln2_b = (const float*)d_in[22];
  const float* fc_w  = (const float*)d_in[23];
  const float* fc_b  = (const float*)d_in[24];
  const float* mproj_w = (const float*)d_in[25];
  const float* mproj_b = (const float*)d_in[26];
  const float* lnf_g = (const float*)d_in[27];
  const float* lnf_b = (const float*)d_in[28];

  // ---- carve workspace ----
  char* wsp = (char*)d_ws;
  auto carve = [&](size_t bytes) -> char* { char* p = wsp; wsp += alignup(bytes); return p; };

  f16* sp_wT    = (f16*)carve((size_t)GD * 128 * 2);
  f16* attn_wT  = (f16*)carve((size_t)GL * 3 * GD * GD * 2);
  f16* proj_wT  = (f16*)carve((size_t)GL * GD * GD * 2);
  f16* fc_wT    = (f16*)carve((size_t)GL * 4 * GD * GD * 2);
  f16* mproj_wT = (f16*)carve((size_t)GL * GD * 4 * GD * 2);
  f16* wh_wT    = (f16*)carve((size_t)GD * GD * 2);
  f16* ws_wT    = (f16*)carve((size_t)GD * GD * 2);
  f16* wd_wT    = (f16*)carve((size_t)GD * GD * 2);

  int*   pre  = (int*)carve((size_t)GM * 4);
  f16*   Sg   = (f16*)carve((size_t)GM * 128 * 2);
  f16*   s16  = (f16*)carve((size_t)GM * GD * 2);
  float* hbuf = (float*)carve((size_t)GM * GD * 4);
  f16*   a16  = (f16*)carve((size_t)GM * GD * 2);
  f16*   o16  = (f16*)carve((size_t)GM * GD * 2);
  f16*   g16  = (f16*)carve((size_t)GM * 4 * GD * 2);  // also aliased as t2/t3 at gate
  float* tmp  = (float*)carve((size_t)GM * 4 * GD * 4);
  f16*   Qh   = (f16*)carve((size_t)GB * GH * GT * 64 * 2);
  f16*   Kh   = (f16*)carve((size_t)GB * GH * GT * 64 * 2);
  f16*   Vth  = (f16*)carve((size_t)GB * GH * 64 * GT * 2);
  float* hf   = (float*)carve((size_t)GB * GD * 4);

  auto ew = [&](size_t n) { return dim3((unsigned)((n + 255) / 256)); };
  auto towT = [&](const float* W, f16* Bt, int K, int N) {
    size_t n = (size_t)K * N;
    k_w_to_f16T<<<ew(n), 256, 0, stream>>>(W, Bt, K, N);
  };
  auto gemm = [&](const f16* A, const f16* Bt, const float* bias, float* C,
                  int Mm, int Nn, int Kk) {
    dim3 g(Nn / 256, Mm / 32);
    k_gemm_wmma<<<g, 128, 0, stream>>>(A, Bt, bias, C, Nn, Kk);
  };

  // ---- 0. convert/transpose all weights to f16 Bt layout ----
  towT(sp_w, sp_wT, 128, GD);
  towT(wh_w, wh_wT, GD, GD);
  towT(ws_w, ws_wT, GD, GD);
  towT(wd_w, wd_wT, GD, GD);
  for (int l = 0; l < GL; ++l) {
    towT(attn_w  + (size_t)l * GD * 3 * GD, attn_wT  + (size_t)l * 3 * GD * GD, GD, 3 * GD);
    towT(proj_w  + (size_t)l * GD * GD,     proj_wT  + (size_t)l * GD * GD,     GD, GD);
    towT(fc_w    + (size_t)l * GD * 4 * GD, fc_wT    + (size_t)l * 4 * GD * GD, GD, 4 * GD);
    towT(mproj_w + (size_t)l * 4 * GD * GD, mproj_wT + (size_t)l * GD * 4 * GD, 4 * GD, GD);
  }

  // ---- 1. state scan + state channel s = state_emb[pre] @ sp_w + sp_b ----
  k_scan<<<1, 32, 0, stream>>>(input_ids, mul, pre);
  k_gather_state<<<ew((size_t)GM * 128), 256, 0, stream>>>(pre, state_emb, Sg);
  gemm(Sg, sp_wT, sp_b, tmp, GM, GD, 128);
  k_f32_to_f16<<<ew((size_t)GM * GD), 256, 0, stream>>>(tmp, s16, (size_t)GM * GD);

  // ---- 2. embeddings: h = tok_emb[ids] + s + wpe ----
  k_embed<<<ew((size_t)GM * GD), 256, 0, stream>>>(input_ids, tok_emb, wpe, s16, hbuf);

  // ---- 3. transformer layers ----
  const size_t nMD = (size_t)GM * GD;
  for (int l = 0; l < GL; ++l) {
    // attention
    k_ln_f16<<<GM, 256, 0, stream>>>(hbuf, ln1_g + (size_t)l * GD, ln1_b + (size_t)l * GD, a16);
    gemm(a16, attn_wT + (size_t)l * 3 * GD * GD, attn_b + (size_t)l * 3 * GD, tmp,
         GM, 3 * GD, GD);
    k_qkv_split<<<ew((size_t)GM * 3 * GD), 256, 0, stream>>>(tmp, Qh, Kh, Vth);
    k_flash_attn<<<dim3(GT / 64, GH, GB), 128, 0, stream>>>(Qh, Kh, Vth, o16);
    gemm(o16, proj_wT + (size_t)l * GD * GD, proj_b + (size_t)l * GD, tmp, GM, GD, GD);
    k_resid_add<<<ew(nMD), 256, 0, stream>>>(hbuf, tmp, nMD);
    // mlp
    k_ln_f16<<<GM, 256, 0, stream>>>(hbuf, ln2_g + (size_t)l * GD, ln2_b + (size_t)l * GD, a16);
    gemm(a16, fc_wT + (size_t)l * 4 * GD * GD, fc_b + (size_t)l * 4 * GD, tmp, GM, 4 * GD, GD);
    k_gelu_f16<<<ew((size_t)GM * 4 * GD), 256, 0, stream>>>(tmp, g16, (size_t)GM * 4 * GD);
    gemm(g16, mproj_wT + (size_t)l * GD * 4 * GD, mproj_b + (size_t)l * GD, tmp, GM, GD, 4 * GD);
    k_resid_add<<<ew(nMD), 256, 0, stream>>>(hbuf, tmp, nMD);
    // gated state-fusion hook after layer 8
    if (l == 8) {
      float* t2 = (float*)g16;                 // g16 is dead until next layer
      float* t3 = t2 + (size_t)GM * GD;
      k_f32_to_f16<<<ew(nMD), 256, 0, stream>>>(hbuf, a16, nMD);
      gemm(a16, wh_wT, wh_b, tmp, GM, GD, GD); // t1 = h@wh + wh_b
      gemm(s16, ws_wT, ws_b, t2,  GM, GD, GD); // t2 = s@ws + ws_b
      gemm(s16, wd_wT, wd_b, t3,  GM, GD, GD); // t3 = s@wd + wd_b
      k_gate_fuse<<<ew(nMD), 256, 0, stream>>>(hbuf, tmp, t2, t3, nMD);
    }
  }

  // ---- 4. final LN (last token only) + head ----
  k_lnf_last<<<GB, 256, 0, stream>>>(hbuf, lnf_g, lnf_b, hf);
  k_head<<<1, 512, 0, stream>>>(hf, head_w, head_b, (float*)d_out);
}
